// Attention_27805618275228
// MI455X (gfx1250) — matrix operand order repaired and checked by reading, not compiled, not run on hardware
//
#include <hip/hip_runtime.h>
#include <math.h>

// ---------------------------------------------------------------------------
// Causal MHA for MI455X (gfx1250): f16 WMMA everywhere, f32 accumulation.
//   B=2, N=2048, DIM=1024, H=16, Dh=64, inner=1024.
// Pipeline:
//   1) convert x -> f16, convert+transpose Wq/Wkv/Wout -> f16 [N,K]
//   2) GEMM (WMMA) : Q  = x @ Wq        [4096,1024]
//   3) GEMM (WMMA) : KV = x @ Wkv ; K -> [4096,1024], V -> vT[b,h,d,j]
//   4) flash attention (WMMA QK^T + WMMA P@V, online softmax, causal)
//   5) GEMM (WMMA) : out = attn @ Wout + bias  (f32 out)
// GEMM staging: CDNA5 async-to-LDS (ASYNCcnt) with LDS double-buffering so
// the DMA of tile k+1 overlaps the WMMAs of tile k.
// ---------------------------------------------------------------------------

typedef __attribute__((ext_vector_type(16))) _Float16 v16h;
typedef __attribute__((ext_vector_type(8)))  _Float16 v8h;
typedef __attribute__((ext_vector_type(8)))  float    v8f;
typedef __attribute__((ext_vector_type(4)))  int      v4i;

#if defined(__has_builtin)
#if __has_builtin(__builtin_amdgcn_global_load_async_to_lds_b128)
#define HAVE_ASYNC_LDS 1
#endif
#endif

// Pointer casts for the async-to-LDS builtin:
//   param 0: v4i in address_space(1) (global), non-const
//   param 1: v4i in address_space(3) (LDS)
#define GPTR_B128(p) ((__attribute__((address_space(1))) v4i*)(p))
#define LPTR_B128(p) ((__attribute__((address_space(3))) v4i*)(p))

static __device__ __forceinline__ void wait_async0() {
#if defined(__has_builtin) && __has_builtin(__builtin_amdgcn_s_wait_asynccnt)
  __builtin_amdgcn_s_wait_asynccnt(0);
#else
  __asm__ volatile("s_wait_asynccnt 0" ::: "memory");
#endif
}

static __device__ __forceinline__ v16h mk16(v8h lo, v8h hi) {
  v16h r;
#pragma unroll
  for (int e = 0; e < 8; ++e) { r[e] = lo[e]; r[e + 8] = hi[e]; }
  return r;
}

static __device__ __forceinline__ v8f wmma_f32_16x16x32(v16h a, v16h b, v8f c) {
  // D = A(16x32 f16) * B(32x16 f16) + C(16x16 f32)
  return __builtin_amdgcn_wmma_f32_16x16x32_f16(false, a, false, b, (short)0, c,
                                                false, false);
}

// ---------------------------------------------------------------------------
// Elementwise converts
// ---------------------------------------------------------------------------
__global__ void cvt_f32_to_f16(const float* __restrict__ in,
                               _Float16* __restrict__ out, int n) {
  int i = blockIdx.x * 256 + threadIdx.x;
  if (i < n) out[i] = (_Float16)in[i];
}

// in: [R][C] f32 row-major  ->  out: [C][R] f16 row-major (transpose)
__global__ void cvt_transpose_f32_to_f16(const float* __restrict__ in,
                                         _Float16* __restrict__ out,
                                         int R, int C) {
  int i = blockIdx.x * 256 + threadIdx.x;
  if (i < R * C) {
    int r = i / C, c = i % C;
    out[(size_t)c * R + r] = (_Float16)in[i];
  }
}

// ---------------------------------------------------------------------------
// WMMA GEMM:  C[M,N] = A[M,K] @ Bt[N,K]^T
// Block 256 thr (8 waves), block tile 128x128, K-step 32, double-buffered LDS.
// Wave grid 4(M) x 2(N); wave tile 32x64 => 2x4 = 8 accumulators.
// mode 0: f16 out [M,N].  mode 1: f32 out + bias[n].
// mode 2: KV split: cols <1024 -> Cv as K [M,1024] f16;
//                   cols >=1024 -> Cv2 as vT[b,h,d,j] f16 (j contiguous).
// ---------------------------------------------------------------------------
__global__ __launch_bounds__(256) void gemm_nt_wmma(
    const _Float16* __restrict__ A,    // [M,K]
    const _Float16* __restrict__ Bt,   // [N,K]
    void* __restrict__ Cv,
    void* __restrict__ Cv2,
    const float* __restrict__ bias,
    int M, int N, int K, int mode) {
  __shared__ __align__(16) _Float16 As[2][128][32];   // 16 KB
  __shared__ __align__(16) _Float16 Bs[2][128][32];   // 16 KB

  const int tid  = threadIdx.x;
  const int lane = tid & 31;
  const int wave = tid >> 5;
  const int bm   = blockIdx.x * 128;
  const int bn   = blockIdx.y * 128;
  const int wm   = (wave >> 1) * 32;
  const int wn   = (wave & 1) * 64;
  const int mrow = lane & 15;
  const int ks   = (lane >> 4) * 8;    // A-frag K sub-offset
  const int kb   = (lane >> 4) * 16;   // unused (B-frag also uses ks)
  const int mb   = (lane >> 4) * 8;    // C-frag row base

  v8f acc[2][4] = {};

  const int lrow = tid >> 1;           // 0..127
  const int lcol = (tid & 1) * 16;     // 0 or 16

  const _Float16* srcA0 = A  + (size_t)(bm + lrow) * K + lcol;
  const _Float16* srcB0 = Bt + (size_t)(bn + lrow) * K + lcol;

  // Stage tile (k0) into LDS buffer `buf`.
  auto stage = [&](int k0, int buf) {
    const _Float16* srcA = srcA0 + k0;
    const _Float16* srcB = srcB0 + k0;
#ifdef HAVE_ASYNC_LDS
    // CDNA5 async DMA: global -> LDS without VGPR staging (ASYNCcnt).
    __builtin_amdgcn_global_load_async_to_lds_b128(
        GPTR_B128(srcA),     LPTR_B128(&As[buf][lrow][lcol]),     0, 0);
    __builtin_amdgcn_global_load_async_to_lds_b128(
        GPTR_B128(srcA + 8), LPTR_B128(&As[buf][lrow][lcol + 8]), 0, 0);
    __builtin_amdgcn_global_load_async_to_lds_b128(
        GPTR_B128(srcB),     LPTR_B128(&Bs[buf][lrow][lcol]),     0, 0);
    __builtin_amdgcn_global_load_async_to_lds_b128(
        GPTR_B128(srcB + 8), LPTR_B128(&Bs[buf][lrow][lcol + 8]), 0, 0);
#else
    v8h a0 = *(const v8h*)srcA;
    v8h a1 = *(const v8h*)(srcA + 8);
    v8h b0 = *(const v8h*)srcB;
    v8h b1 = *(const v8h*)(srcB + 8);
    *(v8h*)&As[buf][lrow][lcol]     = a0;
    *(v8h*)&As[buf][lrow][lcol + 8] = a1;
    *(v8h*)&Bs[buf][lrow][lcol]     = b0;
    *(v8h*)&Bs[buf][lrow][lcol + 8] = b1;
#endif
    if (k0 + 32 < K) {  // prefetch the tile after this one (global_prefetch_b8)
      __builtin_prefetch(srcA + 32, 0, 3);
      __builtin_prefetch(srcB + 32, 0, 3);
    }
  };

  // Prologue: stage tile 0, wait, sync.
  stage(0, 0);
  wait_async0();
  __syncthreads();

  for (int k0 = 0; k0 < K; k0 += 32) {
    const int cur = (k0 >> 5) & 1;

    // Kick off the DMA for the next tile into the other buffer; it runs
    // underneath the WMMAs below (that's what ASYNCcnt buys us).
    if (k0 + 32 < K) stage(k0 + 32, cur ^ 1);

    v16h af[2];
#pragma unroll
    for (int i = 0; i < 2; ++i) {
      const int r = wm + i * 16 + mrow;
      af[i] = mk16(*(const v8h*)&As[cur][r][ks],
                   *(const v8h*)&As[cur][r][16 + ks]);
    }
#pragma unroll
    for (int j = 0; j < 4; ++j) {
      const int r = wn + j * 16 + mrow;
      v16h bf = mk16(*(const v8h*)&Bs[cur][r][ks],
                     *(const v8h*)&Bs[cur][r][16 + ks]);
#pragma unroll
      for (int i = 0; i < 2; ++i)
        acc[i][j] = wmma_f32_16x16x32(af[i], bf, acc[i][j]);
    }

    // Next tile's DMA must be complete (own wave), then block-wide sync
    // (also orders this tile's LDS reads before anyone restages over it).
    wait_async0();
    __syncthreads();
  }

#pragma unroll
  for (int i = 0; i < 2; ++i)
#pragma unroll
    for (int j = 0; j < 4; ++j) {
      const int ccol = bn + wn + j * 16 + mrow;
#pragma unroll
      for (int e = 0; e < 8; ++e) {
        const int crow = bm + wm + i * 16 + mb + e;
        const float v = acc[i][j][e];
        if (mode == 0) {
          ((_Float16*)Cv)[(size_t)crow * N + ccol] = (_Float16)v;
        } else if (mode == 1) {
          ((float*)Cv)[(size_t)crow * N + ccol] = v + bias[ccol];
        } else {  // mode 2: KV split (N tile is entirely K or entirely V)
          if (ccol < 1024) {
            ((_Float16*)Cv)[(size_t)crow * 1024 + ccol] = (_Float16)v;
          } else {
            const int hv = (ccol - 1024) >> 6;   // head
            const int dv = (ccol - 1024) & 63;   // dim within head
            const int bb = crow >> 11;           // batch
            const int jj = crow & 2047;          // sequence pos
            ((_Float16*)Cv2)[((size_t)(bb * 16 + hv) * 64 + dv) * 2048 + jj] =
                (_Float16)v;
          }
        }
      }
    }
}

// ---------------------------------------------------------------------------
// Causal flash attention. Grid (N/64, B*H), block 128 thr = 4 waves.
// Each wave owns a 16-row query tile; key tiles are 32 wide.
//   qh : [B*N, 1024] f16  (head h at cols h*64..h*64+63)
//   kh : [B*N, 1024] f16  (K, same layout as qh)
//   vT : [B*H][64][2048] f16  (V transposed: d-major, j contiguous)
//   oh : [B*N, 1024] f16
// ---------------------------------------------------------------------------
__global__ __launch_bounds__(128) void attn_fmha_wmma(
    const _Float16* __restrict__ qh,
    const _Float16* __restrict__ kh,
    const _Float16* __restrict__ vT,
    _Float16* __restrict__ oh) {
  __shared__ __align__(16) _Float16 plds[4][16][32];  // per-wave P staging

  const int lane = threadIdx.x & 31;
  const int wave = threadIdx.x >> 5;
  const int b    = blockIdx.y >> 4;
  const int h    = blockIdx.y & 15;
  const int q0   = blockIdx.x * 64 + wave * 16;
  const int mrow = lane & 15;
  const int ks   = (lane >> 4) * 8;
  const int kb   = (lane >> 4) * 16;
  const int mb   = (lane >> 4) * 8;
  // SCALE = 64^-0.5 = 0.125 ; fold log2(e) so we can use exp2f
  const float sscale = 0.125f * 1.44269504088896340736f;

  // Q fragments (16 x 64 = two 16x32 A-frags), loaded once
  const _Float16* qp = qh + (size_t)(b * 2048 + q0 + mrow) * 1024 + h * 64;
  v16h qf[2];
#pragma unroll
  for (int c = 0; c < 2; ++c)
    qf[c] = mk16(*(const v8h*)(qp + c * 32 + ks),
                 *(const v8h*)(qp + c * 32 + 16 + ks));

  const _Float16* vTh = vT + (size_t)(b * 16 + h) * 64 * 2048;

  float mi[8], li[8];
#pragma unroll
  for (int i = 0; i < 8; ++i) { mi[i] = -INFINITY; li[i] = 0.f; }
  v8f o[4] = {};  // 16 x 64 output accumulator (four 16x16 C-frags)

  const int jend = q0 + 16;  // keys in [0, jend)
  for (int j0 = 0; j0 < jend; j0 += 32) {
    // ---- S = Q @ K^T : 16x32 as two 16x16 C-fragments -------------------
    v8f s[2];
#pragma unroll
    for (int t = 0; t < 2; ++t) {
      // B-frag: col n = key (j0 + t*16 + mrow); K dim = d (contiguous)
      const _Float16* kp =
          kh + (size_t)(b * 2048 + j0 + t * 16 + mrow) * 1024 + h * 64;
      v8f a = {};
#pragma unroll
      for (int c = 0; c < 2; ++c) {
        v16h kf = mk16(*(const v8h*)(kp + c * 32 + ks),
                       *(const v8h*)(kp + c * 32 + 16 + ks));
        a = wmma_f32_16x16x32(qf[c], kf, a);
      }
      s[t] = a;
    }

    // ---- causal mask + scale, tile row-max ------------------------------
    float mnew[8], rs[8];
#pragma unroll
    for (int i = 0; i < 8; ++i) {
      const int row = q0 + mb + i;
      float s0 = (j0 + mrow)      <= row ? (float)s[0][i] * sscale : -INFINITY;
      float s1 = (j0 + 16 + mrow) <= row ? (float)s[1][i] * sscale : -INFINITY;
      s[0][i] = s0; s[1][i] = s1;
      mnew[i] = fmaxf(s0, s1);
    }
#pragma unroll
    for (int off = 1; off < 16; off <<= 1)
#pragma unroll
      for (int i = 0; i < 8; ++i)
        mnew[i] = fmaxf(mnew[i], __shfl_xor(mnew[i], off, 32));

    // ---- online softmax update; stage P to LDS in A-frag-friendly layout
    float alpha[8];
#pragma unroll
    for (int i = 0; i < 8; ++i) {
      const float mn = fmaxf(mi[i], mnew[i]);
      alpha[i] = exp2f(mi[i] - mn);
      mi[i] = mn;
      const float p0 = exp2f(s[0][i] - mn);
      const float p1 = exp2f(s[1][i] - mn);
      rs[i] = p0 + p1;
      plds[wave][mb + i][mrow]      = (_Float16)p0;
      plds[wave][mb + i][16 + mrow] = (_Float16)p1;
    }
#pragma unroll
    for (int off = 1; off < 16; off <<= 1)
#pragma unroll
      for (int i = 0; i < 8; ++i)
        rs[i] += __shfl_xor(rs[i], off, 32);
#pragma unroll
    for (int i = 0; i < 8; ++i) li[i] = li[i] * alpha[i] + rs[i];
#pragma unroll
    for (int f = 0; f < 4; ++f)
#pragma unroll
      for (int i = 0; i < 8; ++i) o[f][i] *= alpha[i];

    // Same-wave LDS RAW: DS ops are in-order per wave; wait for the stores.
    // (No __syncthreads: per-wave causal trip counts differ within a block.)
    __asm__ volatile("s_wait_dscnt 0" ::: "memory");

    // ---- read P as 16x32 A-fragment -------------------------------------
    v16h pf = mk16(*(const v8h*)&plds[wave][mrow][ks],
                   *(const v8h*)&plds[wave][mrow][16 + ks]);

    // ---- O += P @ V : four 16x16 WMMAs (d chunks of 16) -----------------
    // V B-frag: col n = d (f*16 + mrow), K dim = key j (contiguous in vT).
#pragma unroll
    for (int f = 0; f < 4; ++f) {
      const _Float16* vp = vTh + (size_t)(f * 16 + mrow) * 2048 + j0 + ks;
      v16h vf = mk16(*(const v8h*)vp, *(const v8h*)(vp + 16));
      o[f] = wmma_f32_16x16x32(pf, vf, o[f]);
    }
  }

  // ---- normalize and store ----------------------------------------------
  _Float16* op = oh + (size_t)(b * 2048 + q0) * 1024 + h * 64;
#pragma unroll
  for (int f = 0; f < 4; ++f)
#pragma unroll
    for (int i = 0; i < 8; ++i)
      op[(size_t)(mb + i) * 1024 + f * 16 + mrow] = (_Float16)(o[f][i] / li[i]);
}

// ---------------------------------------------------------------------------
extern "C" void kernel_launch(void* const* d_in, const int* in_sizes, int n_in,
                              void* d_out, int out_size, void* d_ws,
                              size_t ws_size, hipStream_t stream) {
  const float* x    = (const float*)d_in[0];  // [2,2048,1024]
  const float* Wq   = (const float*)d_in[1];  // [1024,1024]
  const float* Wkv  = (const float*)d_in[2];  // [1024,2048]
  const float* Wout = (const float*)d_in[3];  // [1024,1024]
  const float* bout = (const float*)d_in[4];  // [1024]

  const int M = 2 * 2048;  // 4096 rows
  const int D = 1024;

  // Workspace layout (48 MB total)
  char* ws = (char*)d_ws;
  _Float16* xh    = (_Float16*)ws; ws += (size_t)M * D * 2;      // 8 MB
  _Float16* WqT   = (_Float16*)ws; ws += (size_t)D * D * 2;      // 2 MB
  _Float16* WkvT  = (_Float16*)ws; ws += (size_t)2 * D * D * 2;  // 4 MB
  _Float16* WoutT = (_Float16*)ws; ws += (size_t)D * D * 2;      // 2 MB
  _Float16* qhb   = (_Float16*)ws; ws += (size_t)M * D * 2;      // 8 MB
  _Float16* khb   = (_Float16*)ws; ws += (size_t)M * D * 2;      // 8 MB
  _Float16* vTb   = (_Float16*)ws; ws += (size_t)M * D * 2;      // 8 MB
  _Float16* ohb   = (_Float16*)ws; ws += (size_t)M * D * 2;      // 8 MB

  // 1) precision conversion + weight transposes
  {
    int n = M * D;
    cvt_f32_to_f16<<<(n + 255) / 256, 256, 0, stream>>>(x, xh, n);
    cvt_transpose_f32_to_f16<<<(D * D + 255) / 256, 256, 0, stream>>>(
        Wq, WqT, D, D);
    cvt_transpose_f32_to_f16<<<(D * 2 * D + 255) / 256, 256, 0, stream>>>(
        Wkv, WkvT, D, 2 * D);
    cvt_transpose_f32_to_f16<<<(D * D + 255) / 256, 256, 0, stream>>>(
        Wout, WoutT, D, D);
  }

  // 2) Q = x @ Wq   (f16 out)
  gemm_nt_wmma<<<dim3(M / 128, D / 128), 256, 0, stream>>>(
      xh, WqT, (void*)qhb, nullptr, nullptr, M, D, D, 0);

  // 3) KV = x @ Wkv ; K -> khb, V -> vTb (per-head transposed)
  gemm_nt_wmma<<<dim3(M / 128, (2 * D) / 128), 256, 0, stream>>>(
      xh, WkvT, (void*)khb, (void*)vTb, nullptr, M, 2 * D, D, 2);

  // 4) causal flash attention
  attn_fmha_wmma<<<dim3(2048 / 64, 2 * 16), 128, 0, stream>>>(qhb, khb, vTb,
                                                              ohb);

  // 5) out = attn @ Wout + bias (f32 out)
  gemm_nt_wmma<<<dim3(M / 128, D / 128), 256, 0, stream>>>(
      ohb, WoutT, d_out, nullptr, bout, M, D, D, 1);
}
